// FineTune_Biaffine_47064251630002
// MI455X (gfx1250) — matrix operand order repaired and checked by reading, not compiled, not run on hardware
//
#include <hip/hip_runtime.h>
#include <stdint.h>

// Problem constants (from reference setup_inputs)
#define BB 48
#define LL 128
#define DD 1024
#define HH 150          // ffnn out features
#define HP 160          // padded 151 -> 160 (bias col at 150, zeros above)
#define OO 9
#define NIDX (LL * LL)  // 16384
#define MROWS (BB * LL) // 6144

typedef __attribute__((ext_vector_type(16))) __bf16 v16bf;
typedef __attribute__((ext_vector_type(8)))  float  v8f;
typedef __attribute__((ext_vector_type(4)))  unsigned int u32x4;
typedef __attribute__((ext_vector_type(8)))  int i32x8;
typedef __attribute__((ext_vector_type(4)))  int i32x4;

union B16Frag { unsigned int u[8]; v16bf v; };

// round-to-nearest-even f32 -> bf16 (raw u16)
__device__ __forceinline__ unsigned short f2bf(float f) {
  unsigned int u = __builtin_bit_cast(unsigned int, f);
  unsigned int r = u + 0x7FFFu + ((u >> 16) & 1u);
  return (unsigned short)(r >> 16);
}
__device__ __forceinline__ unsigned int packbf(float lo, float hi) {
  return (unsigned int)f2bf(lo) | ((unsigned int)f2bf(hi) << 16);
}
// 16-bit A-matrix 16x32 layout: lanes 0-15 hold K {0..7,16..23}, lanes 16-31 hold K {8..15,24..31}
__device__ __forceinline__ int kpairA(int h, int p) {
  return ((p < 4) ? 0 : 16) + h * 8 + (p & 3) * 2;
}
// 16-bit B-matrix 32x16 layout: lanes 0-15 hold K {0..7,16..23}, lanes 16-31 hold K {8..15,24..31}, 2 K per VGPR
__device__ __forceinline__ int kpairB(int h, int p) {
  return ((p < 4) ? 0 : 16) + h * 8 + (p & 3) * 2;
}

// ---------------------------------------------------------------------------
// Kernel A: S/E projection.  out[m][n] = sum_k bert[m][k] * w[n][k] + bias[n]
// out is bf16 [MROWS][HP]; col 150 = 1.0 (bias_x/bias_y), cols 151..159 = 0.
// ---------------------------------------------------------------------------
__global__ __launch_bounds__(256)
void proj_kernel(const float* __restrict__ bert,
                 const float* __restrict__ w,
                 const float* __restrict__ bias,
                 unsigned short* __restrict__ out) {
  const int mtile = blockIdx.x;
  const int tid = threadIdx.x;
  const int wid = tid >> 5, lane = tid & 31;   // wave32
  const int h = lane >> 4, lm = lane & 15;
  const int Mbase = mtile * 128;

  __shared__ unsigned short Abuf[128 * 32];  // 8 KB, [m][k] bf16
  __shared__ unsigned short Bbuf[HP * 32];   // 10 KB, [n][k] bf16 (k pairs contiguous)

  v8f acc[10];
#pragma unroll
  for (int i = 0; i < 10; ++i) { v8f z = {}; acc[i] = z; }

  for (int k0 = 0; k0 < DD; k0 += 32) {
    // prefetch next A chunk into cache (global_prefetch_b8) while we compute
    if (k0 + 32 < DD)
      __builtin_prefetch(bert + (size_t)(Mbase + (tid >> 1)) * DD + (k0 + 32) + (tid & 1) * 16,
                         0, 1);
    // stage A tile 128x32 (f32 -> bf16), coalesced 64B per thread
    {
      int m = tid >> 1, kb = (tid & 1) * 16;
      const float* src = bert + (size_t)(Mbase + m) * DD + k0 + kb;
      unsigned short* dst = Abuf + m * 32 + kb;
#pragma unroll
      for (int q = 0; q < 16; q += 2)
        *(unsigned int*)(dst + q) = packbf(src[q], src[q + 1]);
    }
    // stage W tile 160x32 (rows >= 150 zero-padded)
    for (int u = tid; u < HP * 2; u += 256) {
      int n = u >> 1, kb = (u & 1) * 16;
      unsigned short* dst = Bbuf + n * 32 + kb;
      if (n < HH) {
        const float* src = w + (size_t)n * DD + k0 + kb;
#pragma unroll
        for (int q = 0; q < 16; q += 2)
          *(unsigned int*)(dst + q) = packbf(src[q], src[q + 1]);
      } else {
#pragma unroll
        for (int q = 0; q < 16; q += 2)
          *(unsigned int*)(dst + q) = 0u;
      }
    }
    __syncthreads();

    B16Frag a;
#pragma unroll
    for (int p = 0; p < 8; ++p)
      a.u[p] = *(const unsigned int*)(Abuf + (wid * 16 + lm) * 32 + kpairA(h, p));
#pragma unroll
    for (int nt = 0; nt < 10; ++nt) {
      B16Frag bf;
#pragma unroll
      for (int p = 0; p < 8; ++p)
        bf.u[p] = *(const unsigned int*)(Bbuf + (nt * 16 + lm) * 32 + kpairB(h, p));
      acc[nt] = __builtin_amdgcn_wmma_f32_16x16x32_bf16(
          false, a.v, false, bf.v, (short)0, acc[nt], false, false);
    }
    __syncthreads();
  }

  // epilogue: bias add, ones column, bf16 store
#pragma unroll
  for (int nt = 0; nt < 10; ++nt) {
#pragma unroll
    for (int r = 0; r < 8; ++r) {
      int m = r + 8 * h;                 // C layout: vgpr r -> row r (+8 for hi half)
      int n = nt * 16 + lm;              // col striped across lanes
      float v = acc[nt][r];
      if (n < HH)      v += bias[n];
      else             v = (n == HH) ? 1.0f : 0.0f;
      out[(size_t)(Mbase + wid * 16 + m) * HP + n] = f2bf(v);
    }
  }
}

// ---------------------------------------------------------------------------
// Kernel B: biaffine per (b,o).  U = X_b @ W_o  then  S = U @ Y_b^T
// Phase-2 Y tile staged via Tensor Data Mover (global -> LDS DMA).
// ---------------------------------------------------------------------------
__global__ __launch_bounds__(256)
void biaffine_kernel(const unsigned short* __restrict__ xws,
                     const unsigned short* __restrict__ yws,
                     const float* __restrict__ wb,   // [9][151][151]
                     float* __restrict__ score) {    // [B][L][L][9]
  const int b = blockIdx.x;
  const int o = blockIdx.y;
  const int tid = threadIdx.x;
  const int wid = tid >> 5, lane = tid & 31;
  const int h = lane >> 4, lm = lane & 15;

  __shared__ unsigned short Ubuf[128 * HP];  // 40 KB
  __shared__ unsigned short SBuf[64 * HP];   // 20 KB: phase1 W-tile (10 KB), phase2 Y half-tile

  // ---- phase 1: U[x][j] = sum_i X[x][i] * W[o][i][j]
  v8f acc1[10];
#pragma unroll
  for (int i = 0; i < 10; ++i) { v8f z = {}; acc1[i] = z; }
  const int xrow0 = b * LL + wid * 16 + lm;

  for (int i0 = 0; i0 < HP; i0 += 32) {
    // stage W[o][i0+i][j] transposed -> SBuf[j][i] bf16 (pad i>=151, j>=151 with 0)
    for (int u = tid; u < 32 * HP; u += 256) {
      int i = u / HP, j = u % HP;
      int gi = i0 + i;
      float v = (gi < 151 && j < 151) ? wb[((size_t)o * 151 + gi) * 151 + j] : 0.0f;
      SBuf[j * 32 + i] = f2bf(v);
    }
    __syncthreads();

    B16Frag a;
#pragma unroll
    for (int p = 0; p < 8; ++p)  // X is already bf16, pairs contiguous in global
      a.u[p] = *(const unsigned int*)(xws + (size_t)xrow0 * HP + i0 + kpairA(h, p));
#pragma unroll
    for (int nt = 0; nt < 10; ++nt) {
      B16Frag bf;
#pragma unroll
      for (int p = 0; p < 8; ++p)
        bf.u[p] = *(const unsigned int*)(SBuf + (nt * 16 + lm) * 32 + kpairB(h, p));
      acc1[nt] = __builtin_amdgcn_wmma_f32_16x16x32_bf16(
          false, a.v, false, bf.v, (short)0, acc1[nt], false, false);
    }
    __syncthreads();
  }
  // spill U to LDS as bf16 (re-layout C frag -> A frag via LDS bounce)
#pragma unroll
  for (int nt = 0; nt < 10; ++nt)
#pragma unroll
    for (int r = 0; r < 8; ++r) {
      int m = wid * 16 + r + 8 * h;
      int n = nt * 16 + lm;
      Ubuf[m * HP + n] = f2bf(acc1[nt][r]);
    }
  __syncthreads();

  // ---- phase 2: score[x][y] = sum_j U[x][j] * Y[y][j], y in halves of 64
  for (int half = 0; half < 2; ++half) {
    // TDM: DMA the 64x160 bf16 Y half-tile (20480 B, contiguous) into SBuf.
    // Issued by wave 0 only (TDM ignores EXEC; scalar branch avoids 8x issue).
    if (wid == 0) {
      const unsigned TLEN = 64 * HP;  // 10240 elements of 2 bytes
      unsigned ldsbase = (unsigned)(uintptr_t)(&SBuf[0]);
      unsigned long long ga =
          (unsigned long long)(uintptr_t)(yws + ((size_t)b * LL + half * 64) * HP);
      u32x4 g0;
      g0[0] = 1u;                                            // count=1, user mode
      g0[1] = ldsbase;                                       // lds_addr
      g0[2] = (unsigned)(ga & 0xFFFFFFFFu);                  // global_addr[31:0]
      g0[3] = (unsigned)((ga >> 32) & 0x1FFFFFFu) | (2u << 30);  // addr[56:32] | type=2
      i32x8 g1;
      g1[0] = (int)(1u << 16);                               // data_size=2B, mask=0
      g1[1] = (int)((TLEN & 0xFFFFu) << 16);                 // tensor_dim0[15:0]
      g1[2] = (int)((TLEN >> 16) | (1u << 16));              // tdim0[31:16] | tensor_dim1=1
      g1[3] = (int)((TLEN & 0xFFFFu) << 16);                 // tile_dim0
      g1[4] = 1;                                             // tile_dim1=1, tile_dim2=0
      g1[5] = (int)TLEN;                                     // tensor_dim0_stride lo
      g1[6] = 0;
      g1[7] = 0;
      i32x4 gz4 = {0, 0, 0, 0};
      i32x8 gz8 = {0, 0, 0, 0, 0, 0, 0, 0};
      __builtin_amdgcn_tensor_load_to_lds(g0, g1, gz4, gz4, gz8, 0);
      __builtin_amdgcn_s_wait_tensorcnt(0);
    }
    __syncthreads();

    v8f acc2[4];
#pragma unroll
    for (int i = 0; i < 4; ++i) { v8f z = {}; acc2[i] = z; }
    for (int k0 = 0; k0 < HP; k0 += 32) {
      B16Frag a;
#pragma unroll
      for (int p = 0; p < 8; ++p)
        a.u[p] = *(const unsigned int*)(Ubuf + (wid * 16 + lm) * HP + k0 + kpairA(h, p));
#pragma unroll
      for (int nt = 0; nt < 4; ++nt) {
        B16Frag bf;
#pragma unroll
        for (int p = 0; p < 8; ++p)  // B[k=j][n=y] = Y[y][j]; pairs contiguous per row
          bf.u[p] = *(const unsigned int*)(SBuf + (nt * 16 + lm) * HP + k0 + kpairB(h, p));
        acc2[nt] = __builtin_amdgcn_wmma_f32_16x16x32_bf16(
            false, a.v, false, bf.v, (short)0, acc2[nt], false, false);
      }
    }
#pragma unroll
    for (int nt = 0; nt < 4; ++nt)
#pragma unroll
      for (int r = 0; r < 8; ++r) {
        int x = wid * 16 + r + 8 * h;
        int y = half * 64 + nt * 16 + lm;
        score[(((size_t)b * LL + x) * LL + y) * OO + o] = acc2[nt][r];
      }
    __syncthreads();
  }
}

// ---------------------------------------------------------------------------
// Kernel C: greedy decode per sentence (argmax -> sortable keys -> bitonic
// sort -> sequential overlap-suppression scan with LDS-staged chunks)
// ---------------------------------------------------------------------------
__global__ __launch_bounds__(256)
void decode_kernel(const float* __restrict__ score,
                   const int* __restrict__ tags,
                   unsigned long long* __restrict__ keys,  // [B][NIDX]
                   int* __restrict__ ansbuf,               // [B][NIDX]
                   int* __restrict__ result) {             // [B][L][L]
  const int s = blockIdx.x;
  const int tid = threadIdx.x;
  unsigned long long* k = keys + (size_t)s * NIDX;
  int* ansv = ansbuf + (size_t)s * NIDX;
  int* res = result + (size_t)s * NIDX;

  // 1) argmax over O (first-max tie rule), build sortable key, init result=1
  for (int idx = tid; idx < NIDX; idx += 256) {
    const float* sc = score + ((size_t)s * NIDX + idx) * OO;
    float best = sc[0];
    int ba = 0;
#pragma unroll
    for (int o = 1; o < OO; ++o) {
      float v = sc[o];
      if (v > best) { best = v; ba = o; }
    }
    bool valid = (ba != 1) && (tags[(size_t)s * NIDX + idx] > 0);
    unsigned int u = __builtin_bit_cast(unsigned int, best);
    unsigned int ord = (u & 0x80000000u) ? ~u : (u | 0x80000000u);  // monotone map
    unsigned int inv = valid ? ~ord : 0xFFFFFFFFu;  // ascending sort => score desc
    k[idx] = ((unsigned long long)inv << 32) | (unsigned int)idx;   // idx breaks ties
    ansv[idx] = ba;
    res[idx] = 1;  // NON_ENTITY
  }
  __syncthreads();

  // 2) bitonic sort, ascending (invalid keys sink to the end)
  for (unsigned int kk = 2; kk <= NIDX; kk <<= 1) {
    for (unsigned int j = kk >> 1; j > 0; j >>= 1) {
      for (unsigned int i = tid; i < NIDX; i += 256) {
        unsigned int ixj = i ^ j;
        if (ixj > i) {
          unsigned long long a = k[i], b2 = k[ixj];
          bool up = ((i & kk) == 0);
          if ((a > b2) == up) { k[i] = b2; k[ixj] = a; }
        }
      }
      __syncthreads();
    }
  }

  // 3) sequential greedy scan on thread 0; keys prefetched in 4K LDS chunks
  __shared__ unsigned long long skeys[4096];  // 32 KB
  __shared__ int sans[4096];                  // 16 KB
  __shared__ int sdone;
  __shared__ unsigned int msk[8];
  if (tid == 0) {
    sdone = 0;
    for (int z = 0; z < 8; ++z) msk[z] = 0u;
  }
  __syncthreads();

  for (int c = 0; c < NIDX / 4096; ++c) {
    if (sdone) break;
    for (int q = tid; q < 4096; q += 256) {
      unsigned long long kv = k[c * 4096 + q];
      skeys[q] = kv;
      sans[q] = ansv[kv & 0xFFFFu];
    }
    __syncthreads();
    if (tid == 0) {
      unsigned int st0 = msk[0], st1 = msk[1], st2 = msk[2], st3 = msk[3];
      unsigned int cv0 = msk[4], cv1 = msk[5], cv2 = msk[6], cv3 = msk[7];
      for (int q = 0; q < 4096; ++q) {
        unsigned long long kv = skeys[q];
        unsigned int inv = (unsigned int)(kv >> 32);
        if (inv == 0xFFFFFFFFu) { sdone = 1; break; }  // rest are invalid
        int idx = (int)(kv & 0xFFFFu);
        int i = idx >> 7, j = idx & 127;
        unsigned int m0 = 0, m1 = 0, m2 = 0, m3 = 0;
        if (j >= i) {  // span [i, j]; empty when j < i
          auto spanWord = [&](int w0) -> unsigned int {
            int lo = w0 * 32, hi = lo + 31;
            int a = (i > lo) ? i : lo;
            int b3 = (j < hi) ? j : hi;
            if (b3 < a) return 0u;
            int len = b3 - a + 1;
            unsigned int mf = (len >= 32) ? 0xFFFFFFFFu : ((1u << len) - 1u);
            return mf << (a - lo);
          };
          m0 = spanWord(0); m1 = spanWord(1); m2 = spanWord(2); m3 = spanWord(3);
        }
        unsigned int conflict = (st0 & m0) | (st1 & m1) | (st2 & m2) | (st3 & m3);
        unsigned int cw = (i < 32) ? cv0 : (i < 64) ? cv1 : (i < 96) ? cv2 : cv3;
        unsigned int covbit = (cw >> (i & 31)) & 1u;
        if (!conflict && !covbit) {
          if (i < 32)      st0 |= 1u << i;
          else if (i < 64) st1 |= 1u << (i - 32);
          else if (i < 96) st2 |= 1u << (i - 64);
          else             st3 |= 1u << (i - 96);
          cv0 |= m0; cv1 |= m1; cv2 |= m2; cv3 |= m3;
          res[idx] = sans[q];
        }
      }
      msk[0] = st0; msk[1] = st1; msk[2] = st2; msk[3] = st3;
      msk[4] = cv0; msk[5] = cv1; msk[6] = cv2; msk[7] = cv3;
    }
    __syncthreads();
  }
}

// ---------------------------------------------------------------------------
extern "C" void kernel_launch(void* const* d_in, const int* in_sizes, int n_in,
                              void* d_out, int out_size, void* d_ws, size_t ws_size,
                              hipStream_t stream) {
  (void)in_sizes; (void)n_in; (void)out_size; (void)ws_size;
  const float* bert    = (const float*)d_in[0];
  const int*   tags    = (const int*)d_in[1];
  const float* w_start = (const float*)d_in[2];
  const float* b_start = (const float*)d_in[3];
  const float* w_end   = (const float*)d_in[4];
  const float* b_end   = (const float*)d_in[5];
  const float* w_biaf  = (const float*)d_in[6];

  float* score = (float*)d_out;                                   // [B,L,L,O] f32
  int* result = (int*)(score + (size_t)BB * LL * LL * OO);        // [B,L,L] i32

  unsigned char* ws = (unsigned char*)d_ws;
  unsigned short* xws = (unsigned short*)ws;                      // [MROWS][HP] bf16
  unsigned short* yws = xws + (size_t)MROWS * HP;
  unsigned long long* keys = (unsigned long long*)(yws + (size_t)MROWS * HP);
  int* ansbuf = (int*)(keys + (size_t)BB * NIDX);

  proj_kernel<<<dim3(MROWS / 128), 256, 0, stream>>>(bert, w_start, b_start, xws);
  proj_kernel<<<dim3(MROWS / 128), 256, 0, stream>>>(bert, w_end, b_end, yws);
  biaffine_kernel<<<dim3(BB, OO), 256, 0, stream>>>(xws, yws, w_biaf, score);
  decode_kernel<<<dim3(BB), 256, 0, stream>>>(score, tags, keys, ansbuf, result);
}
